// SimpleHackAttention_7189775253985
// MI455X (gfx1250) — compile-verified, matrix operand order checked by reading, not run on hardware
//
#include <hip/hip_runtime.h>

typedef __attribute__((ext_vector_type(16))) __bf16 v16bf;
typedef __attribute__((ext_vector_type(8)))  float  v8f;
typedef __attribute__((ext_vector_type(4)))  unsigned int v4u;
typedef __attribute__((ext_vector_type(8)))  int v8i_;
typedef __attribute__((ext_vector_type(4)))  int v4i_;

// ---------------- problem sizes ----------------
static constexpr int       Bsz   = 4;
static constexpr int       Ntok  = 8192;
static constexpr int       Dmod  = 256;
static constexpr int       Hn    = 8;
static constexpr int       HDd   = 64;
static constexpr int       INNER = 512;
static constexpr int       QKV3  = 1536;
static constexpr int       Mland = 256;
static constexpr int       Lgrp  = 32;
static constexpr int       NCLS  = 1000;
static constexpr int       NCLSP = 1024;
static constexpr int       BHn   = 32;
static constexpr float     SCALE = 0.125f;  // 64^-0.5

// ---------------- bf16 helpers (raw u16 storage) ----------------
static __device__ __forceinline__ float bf2f(unsigned short u) {
  union { unsigned int i; float f; } v; v.i = ((unsigned int)u) << 16; return v.f;
}
static __device__ __forceinline__ unsigned short f2bf(float f) {
  union { float f; unsigned int i; } v; v.f = f;
  unsigned int r = v.i + 0x7FFFu + ((v.i >> 16) & 1u);
  return (unsigned short)(r >> 16);
}

struct FragBF16 { union { v16bf v; uint4 q[2]; unsigned short u[16]; }; };

// ---------------------------------------------------------------------
// TDM: issue a 2-D bf16 tile DMA  global -> LDS  (Tensor Data Mover).
// D# group0: count=1 | lds_addr | global_addr(57b) | type=2.
// D# group1: data_size=1(2B); tensor dims huge (tiles are always in
// bounds, so disable clipping); tile_dim0 x tile_dim1; dim0 stride = ld.
// Groups 2/3 zero (2-D tensor). Tracked by TENSORcnt.
// This toolchain exposes the 6-arg builtin:
//   (uint32x4 g0, int32x8 g1, int32x4 g2, int32x4 g3, int32x8 pad, i32 cpol)
// ---------------------------------------------------------------------
static __device__ __forceinline__ void tdm_load_tile(
    const unsigned short* gptr, unsigned int lds_off, long long ld_elems,
    unsigned int tile0, unsigned int tile1) {
  unsigned long long ga = (unsigned long long)(uintptr_t)gptr;
  unsigned long long s0 = (unsigned long long)ld_elems;  // dim0 stride (elems)
  const unsigned int td0 = 0x7FFFFFFFu, td1 = 0x7FFFFFFFu;
  v4u g0;
  g0.x = 1u;                                        // count=1 (valid user D#)
  g0.y = lds_off;                                   // lds_addr (bytes)
  g0.z = (unsigned int)(ga & 0xFFFFFFFFu);          // global_addr[31:0]
  g0.w = (unsigned int)((ga >> 32) & 0x01FFFFFFu)   // global_addr[56:32]
         | (2u << 30);                              // type=2 ("image")
  v8i_ g1;
  g1[0] = (int)(1u << 16);                          // data_size=1 -> 2 bytes
  g1[1] = (int)((td0 & 0xFFFFu) << 16);             // tensor_dim0 lo
  g1[2] = (int)((td0 >> 16) | ((td1 & 0xFFFFu) << 16));
  g1[3] = (int)((td1 >> 16) | (tile0 << 16));       // tile_dim0
  g1[4] = (int)(tile1);                             // tile_dim1 (tile_dim2=0)
  g1[5] = (int)(s0 & 0xFFFFFFFFu);                  // dim0_stride lo
  g1[6] = (int)((s0 >> 32) & 0xFFFFu);              // dim0_stride hi
  g1[7] = 0;
  v4i_ zz4 = {0, 0, 0, 0};
  v8i_ zz8 = {0, 0, 0, 0, 0, 0, 0, 0};
  __builtin_amdgcn_tensor_load_to_lds(g0, g1, zz4, zz4, zz8, 0);
}

// =====================================================================
// Generic batched WMMA GEMM, TDM-fed & LDS double-buffered.
//   C[M,N] (+batch) = A[M,K] * op(B)     (fp32 accumulate)
//   TRANS_B=true : B stored [N,K] row-major (dot along contiguous K)
//   TRANS_B=false: B stored [K,N] row-major (tile transposed via LDS)
// Block = 128 threads (4 waves), wave tile 32x32 (2x2 wmma frags),
// block tile 64x64, K stepped by 32. Wave 0 issues one TDM tile DMA for
// A (64x32) and one for B per K-step into the ping-pong LDS buffers,
// waits TENSORcnt, and the next step's DMA overlaps this step's WMMAs.
// M/N params are the *store* bounds (tiles themselves always in-buffer).
// =====================================================================
template <bool TRANS_B, bool OUT_BF16>
__global__ __launch_bounds__(128)
void gemm_wmma(const unsigned short* __restrict__ Ag,
               const unsigned short* __restrict__ Bg,
               void* __restrict__ Cg,
               int M, int N, int K,
               int lda, int ldb, int ldc,
               int hcount,
               long long sAb, long long sAh,
               long long sBb, long long sBh,
               long long sCb, long long sCh) {
  __shared__ __attribute__((aligned(16))) unsigned short smA[2][64 * 32];
  __shared__ __attribute__((aligned(16))) unsigned short smB[2][64 * 32];

  int z  = blockIdx.z;
  int bb = z / hcount, hh = z % hcount;
  const unsigned short* A = Ag + bb * sAb + hh * sAh;
  const unsigned short* B = Bg + bb * sBb + hh * sBh;
  long long coff = bb * sCb + hh * sCh;

  int lane  = threadIdx.x & 31;
  int w     = threadIdx.x >> 5;
  int wr    = w >> 1, wc = w & 1;
  int m0    = blockIdx.y * 64 + wr * 32;
  int n0    = blockIdx.x * 64 + wc * 32;
  int row16 = lane & 15;
  int khalf = lane >> 4;

  const unsigned short* Atile = A + (long long)(blockIdx.y * 64) * lda;
  const unsigned short* Btile =
      TRANS_B ? (B + (long long)(blockIdx.x * 64) * ldb)
              : (B + blockIdx.x * 64);

  auto issue = [&](int buf, int k) {
    if (threadIdx.x < 32) {
      tdm_load_tile(Atile + k, (unsigned int)(uintptr_t)&smA[buf][0],
                    lda, 32u, 64u);
      if (TRANS_B)
        tdm_load_tile(Btile + k, (unsigned int)(uintptr_t)&smB[buf][0],
                      ldb, 32u, 64u);
      else
        tdm_load_tile(Btile + (long long)k * ldb,
                      (unsigned int)(uintptr_t)&smB[buf][0], ldb, 64u, 32u);
    }
  };

  v8f zero = {0.f, 0.f, 0.f, 0.f, 0.f, 0.f, 0.f, 0.f};
  v8f acc[2][2];
  acc[0][0] = zero; acc[0][1] = zero; acc[1][0] = zero; acc[1][1] = zero;

  issue(0, 0);
  int buf = 0;
  for (int k = 0; k < K; k += 32, buf ^= 1) {
    if (threadIdx.x < 32) __builtin_amdgcn_s_wait_tensorcnt(0);
    __syncthreads();                       // tile in LDS, visible to all
    if (k + 32 < K) issue(buf ^ 1, k + 32);  // overlap next DMA with math

    FragBF16 a[2], b[2];
#pragma unroll
    for (int i = 0; i < 2; i++) {
      const unsigned short* ap =
          &smA[buf][(wr * 32 + 16 * i + row16) * 32 + 8 * khalf];
      a[i].q[0] = *(const uint4*)ap;
      a[i].q[1] = *(const uint4*)(ap + 16);
    }
    if (TRANS_B) {
#pragma unroll
      for (int j = 0; j < 2; j++) {
        const unsigned short* bp =
            &smB[buf][(wc * 32 + 16 * j + row16) * 32 + 8 * khalf];
        b[j].q[0] = *(const uint4*)bp;
        b[j].q[1] = *(const uint4*)(bp + 16);
      }
    } else {
#pragma unroll
      for (int j = 0; j < 2; j++) {
        int ncol = wc * 32 + 16 * j + row16;
#pragma unroll
        for (int e = 0; e < 8; e++)
          b[j].u[e] = smB[buf][(8 * khalf + e) * 64 + ncol];
#pragma unroll
        for (int e = 0; e < 8; e++)
          b[j].u[8 + e] = smB[buf][(16 + 8 * khalf + e) * 64 + ncol];
      }
    }
#pragma unroll
    for (int i = 0; i < 2; i++)
#pragma unroll
      for (int j = 0; j < 2; j++)
        acc[i][j] = __builtin_amdgcn_wmma_f32_16x16x32_bf16(
            false, a[i].v, false, b[j].v, (short)0, acc[i][j], false, false);
    __syncthreads();                       // done reading buf before reuse
  }

  int rbase = khalf * 8;
  int ccol  = lane & 15;
#pragma unroll
  for (int i = 0; i < 2; i++)
#pragma unroll
    for (int j = 0; j < 2; j++) {
      v8f av = acc[i][j];
#pragma unroll
      for (int r = 0; r < 8; r++) {
        int rowi = m0 + 16 * i + rbase + r;
        int coli = n0 + 16 * j + ccol;
        if (rowi < M && coli < N) {
          long long idx = coff + (long long)rowi * ldc + coli;
          if (OUT_BF16) ((unsigned short*)Cg)[idx] = f2bf(av[r]);
          else          ((float*)Cg)[idx]          = av[r];
        }
      }
    }
}

// ---------------- elementwise / reduction kernels ----------------
__global__ __launch_bounds__(256)
void f32_to_bf16_pad(const float* __restrict__ in, unsigned short* __restrict__ out,
                     long long nin, long long nout) {
  long long i = (long long)blockIdx.x * 256 + threadIdx.x;
  if (i < nout) out[i] = (i < nin) ? f2bf(in[i]) : (unsigned short)0;
}

// landmark pooling: sum 32 consecutive tokens of q and k (bf16 in/out)
__global__ __launch_bounds__(64)
void landmarks(const unsigned short* __restrict__ qkv,
               unsigned short* __restrict__ qlb, unsigned short* __restrict__ klb) {
  int idx = blockIdx.x;                 // (b*8+h)*256 + mi
  int mi  = idx & 255;
  int hh  = (idx >> 8) & 7;
  int bb  = idx >> 11;
  int dh  = threadIdx.x;
  long long base =
      ((long long)(bb * Ntok + mi * Lgrp)) * QKV3 + hh * HDd + dh;
  float qs = 0.f, ks = 0.f;
  for (int j = 0; j < Lgrp; j++) {
    qs += bf2f(qkv[base + (long long)j * QKV3]);
    ks += bf2f(qkv[base + (long long)j * QKV3 + INNER]);
  }
  long long o = (long long)idx * HDd + dh;
  qlb[o] = f2bf(qs);
  klb[o] = f2bf(ks);
}

// row softmax over bf16 logits*scale -> bf16 (and optional f32) probs
__global__ __launch_bounds__(256)
void softmax_rows(const unsigned short* __restrict__ in,
                  unsigned short* __restrict__ outb,
                  float* __restrict__ outf, int cols, float scale) {
  long long row = blockIdx.x;
  const unsigned short* rin = in + row * cols;
  __shared__ float red[256];
  int tid = threadIdx.x;
  float mx = -3.4e38f;
  for (int i = tid; i < cols; i += 256) mx = fmaxf(mx, bf2f(rin[i]) * scale);
  red[tid] = mx; __syncthreads();
  for (int s = 128; s > 0; s >>= 1) {
    if (tid < s) red[tid] = fmaxf(red[tid], red[tid + s]);
    __syncthreads();
  }
  mx = red[0]; __syncthreads();
  float sum = 0.f;
  for (int i = tid; i < cols; i += 256) sum += __expf(bf2f(rin[i]) * scale - mx);
  red[tid] = sum; __syncthreads();
  for (int s = 128; s > 0; s >>= 1) {
    if (tid < s) red[tid] += red[tid + s];
    __syncthreads();
  }
  float inv = 1.f / red[0];
  for (int i = tid; i < cols; i += 256) {
    float v = __expf(bf2f(rin[i]) * scale - mx) * inv;
    outb[row * cols + i] = f2bf(v);
    if (outf) outf[row * cols + i] = v;
  }
}

__global__ __launch_bounds__(256)
void rowsum_abs(const float* __restrict__ x, float* __restrict__ rs) {
  long long r = blockIdx.x;                       // bh*256 + i
  __shared__ float red[256];
  int tid = threadIdx.x;
  red[tid] = fabsf(x[r * 256 + tid]); __syncthreads();
  for (int s = 128; s > 0; s >>= 1) {
    if (tid < s) red[tid] += red[tid + s];
    __syncthreads();
  }
  if (tid == 0) rs[r] = red[0];
}

__global__ __launch_bounds__(256)
void colsum_abs(const float* __restrict__ x, float* __restrict__ cs) {
  int bh = blockIdx.x >> 8, j = blockIdx.x & 255;
  __shared__ float red[256];
  int tid = threadIdx.x;
  red[tid] = fabsf(x[((long long)bh * 256 + tid) * 256 + j]); __syncthreads();
  for (int s = 128; s > 0; s >>= 1) {
    if (tid < s) red[tid] += red[tid + s];
    __syncthreads();
  }
  if (tid == 0) cs[blockIdx.x] = red[0];
}

__global__ __launch_bounds__(256)
void maxprod(const float* __restrict__ rs, const float* __restrict__ cs,
             float* __restrict__ inv) {
  __shared__ float red[256];
  int tid = threadIdx.x;
  float mr = 0.f, mc = 0.f;
  for (int i = tid; i < BHn * 256; i += 256) {
    mr = fmaxf(mr, rs[i]); mc = fmaxf(mc, cs[i]);
  }
  red[tid] = mr; __syncthreads();
  for (int s = 128; s > 0; s >>= 1) {
    if (tid < s) red[tid] = fmaxf(red[tid], red[tid + s]);
    __syncthreads();
  }
  mr = red[0]; __syncthreads();
  red[tid] = mc; __syncthreads();
  for (int s = 128; s > 0; s >>= 1) {
    if (tid < s) red[tid] = fmaxf(red[tid], red[tid + s]);
    __syncthreads();
  }
  if (tid == 0) inv[0] = 1.f / (mr * red[0]);
}

// z0 = x^T * invscale   (per bh 256x256)
__global__ __launch_bounds__(256)
void pinv_init(const float* __restrict__ x, const float* __restrict__ inv,
               unsigned short* __restrict__ zb) {
  int bh = blockIdx.x >> 8, i = blockIdx.x & 255, j = threadIdx.x;
  float v = x[((long long)bh * 256 + j) * 256 + i] * inv[0];
  zb[(long long)blockIdx.x * 256 + j] = f2bf(v);
}

// Pb = bf16(P) (optional); Tb = bf16(c*I - P)
__global__ __launch_bounds__(256)
void pinv_combine(const float* __restrict__ P, unsigned short* __restrict__ Pb,
                  unsigned short* __restrict__ Tb, float c) {
  int i = blockIdx.x & 255, j = threadIdx.x;
  long long idx = (long long)blockIdx.x * 256 + j;
  float v = P[idx];
  if (Pb) Pb[idx] = f2bf(v);
  Tb[idx] = f2bf(((i == j) ? c : 0.f) - v);
}

__global__ __launch_bounds__(256)
void scale_to_bf16(const float* __restrict__ P, unsigned short* __restrict__ o,
                   float s) {
  long long idx = (long long)blockIdx.x * 256 + threadIdx.x;
  o[idx] = f2bf(P[idx] * s);
}

// out[bh][i][j] = in[bh][j][i]   (bf16 256x256 per bh)
__global__ __launch_bounds__(256)
void transpose256(const unsigned short* __restrict__ in,
                  unsigned short* __restrict__ out) {
  int bh = blockIdx.x >> 8, i = blockIdx.x & 255, j = threadIdx.x;
  out[(long long)blockIdx.x * 256 + j] = in[((long long)bh * 256 + j) * 256 + i];
}

// WT[bh][d][i] = bf16(W[bh][i][d])   W: [32,256,64] f32
__global__ __launch_bounds__(256)
void wtranspose(const float* __restrict__ W, unsigned short* __restrict__ WT) {
  int bh = blockIdx.x >> 6, d = blockIdx.x & 63, i = threadIdx.x;
  WT[(long long)blockIdx.x * 256 + i] =
      f2bf(W[((long long)bh * 256 + i) * 64 + d]);
}

// =====================================================================
extern "C" void kernel_launch(void* const* d_in, const int* in_sizes, int n_in,
                              void* d_out, int out_size, void* d_ws,
                              size_t ws_size, hipStream_t stream) {
  (void)in_sizes; (void)n_in; (void)out_size; (void)ws_size;
  const float* x    = (const float*)d_in[0];
  const float* wqkv = (const float*)d_in[1];
  const float* wout = (const float*)d_in[2];
  float* out = (float*)d_out;

  char* ws = (char*)d_ws;
  size_t off = 0;
  auto alloc = [&](size_t bytes) -> char* {
    char* p = ws + off;
    off = (off + bytes + 255) & ~(size_t)255;
    return p;
  };

  const long long BN   = (long long)Bsz * Ntok;         // 32768
  const long long BND  = BN * Dmod;                     // x elems
  const long long QKVE = BN * QKV3;                     // qkv elems
  const long long LE   = (long long)BHn * Mland * HDd;  // landmark elems
  const long long A2E  = (long long)BHn * Mland * Mland;
  const long long BIGE = (long long)BHn * Ntok * Mland; // 67,108,864

  unsigned short* xb     = (unsigned short*)alloc(2 * BND);
  unsigned short* wqkvb  = (unsigned short*)alloc(2ULL * QKV3 * Dmod);
  unsigned short* woutb  = (unsigned short*)alloc(2ULL * NCLSP * INNER);
  unsigned short* qkvb   = (unsigned short*)alloc(2 * QKVE);
  unsigned short* qlb    = (unsigned short*)alloc(2 * LE);
  unsigned short* klb    = (unsigned short*)alloc(2 * LE);
  unsigned short* sim2b  = (unsigned short*)alloc(2 * A2E);
  float*          attn2f = (float*)alloc(4 * A2E);
  unsigned short* attn2b = (unsigned short*)alloc(2 * A2E);
  float*          rsums  = (float*)alloc(4ULL * BHn * 256);
  float*          csums  = (float*)alloc(4ULL * BHn * 256);
  float*          invsc  = (float*)alloc(256);
  unsigned short* zb     = (unsigned short*)alloc(2 * A2E);
  float*          P0f    = (float*)alloc(4 * A2E);
  float*          P1f    = (float*)alloc(4 * A2E);
  unsigned short* xzb    = (unsigned short*)alloc(2 * A2E);
  unsigned short* tb     = (unsigned short*)alloc(2 * A2E);
  unsigned short* zTb    = (unsigned short*)alloc(2 * A2E);
  unsigned short* big1   = (unsigned short*)alloc(2 * BIGE); // sim1/sim3
  unsigned short* big2   = (unsigned short*)alloc(2 * BIGE); // attn1/attn3
  unsigned short* big3   = (unsigned short*)alloc(2 * BIGE); // T1
  float*          Wf     = (float*)alloc(4 * LE);
  unsigned short* WTb    = (unsigned short*)alloc(2 * LE);
  unsigned short* ohb    = (unsigned short*)alloc(2 * BN * INNER);

  // 1) fp32 -> bf16 conversions (classifier weights zero-padded 1000->1024)
  f32_to_bf16_pad<<<dim3((unsigned)((BND + 255) / 256)), 256, 0, stream>>>(
      x, xb, BND, BND);
  f32_to_bf16_pad<<<dim3((QKV3 * Dmod + 255) / 256), 256, 0, stream>>>(
      wqkv, wqkvb, (long long)QKV3 * Dmod, (long long)QKV3 * Dmod);
  f32_to_bf16_pad<<<dim3((NCLSP * INNER + 255) / 256), 256, 0, stream>>>(
      wout, woutb, (long long)NCLS * INNER, (long long)NCLSP * INNER);

  // 2) qkv = x @ wqkv^T   [32768,1536]
  gemm_wmma<true, true><<<dim3(QKV3 / 64, (unsigned)(BN / 64), 1), 128, 0, stream>>>(
      xb, wqkvb, qkvb, (int)BN, QKV3, Dmod, Dmod, Dmod, QKV3,
      1, 0, 0, 0, 0, 0, 0);

  // 3) landmark pooling
  landmarks<<<dim3(BHn * Mland), 64, 0, stream>>>(qkvb, qlb, klb);

  // 4) sim2 = ql @ kl^T  (batch 32)
  gemm_wmma<true, true><<<dim3(4, 4, BHn), 128, 0, stream>>>(
      qlb, klb, sim2b, Mland, Mland, HDd, HDd, HDd, Mland,
      1, (long long)Mland * HDd, 0, (long long)Mland * HDd, 0,
      (long long)Mland * Mland, 0);

  // 5) attn2 = softmax(sim2 * scale) -> f32 + bf16
  softmax_rows<<<dim3(BHn * Mland), 256, 0, stream>>>(sim2b, attn2b, attn2f,
                                                      Mland, SCALE);

  // 6) pinv init: z = attn2^T / (max(colsum)*max(rowsum))
  rowsum_abs<<<dim3(BHn * 256), 256, 0, stream>>>(attn2f, rsums);
  colsum_abs<<<dim3(BHn * 256), 256, 0, stream>>>(attn2f, csums);
  maxprod<<<dim3(1), 256, 0, stream>>>(rsums, csums, invsc);
  pinv_init<<<dim3(BHn * 256), 256, 0, stream>>>(attn2f, invsc, zb);

  // 7) 6 Newton-Schulz iterations: z = 0.25 z (13I - xz(15I - xz(7I - xz)))
  const long long sq = (long long)Mland * Mland;
  for (int it = 0; it < 6; it++) {
    gemm_wmma<false, false><<<dim3(4, 4, BHn), 128, 0, stream>>>(
        attn2b, zb, P0f, Mland, Mland, Mland, Mland, Mland, Mland,
        1, sq, 0, sq, 0, sq, 0);                       // xz = x @ z
    pinv_combine<<<dim3(BHn * 256), 256, 0, stream>>>(P0f, xzb, tb, 7.f);
    gemm_wmma<false, false><<<dim3(4, 4, BHn), 128, 0, stream>>>(
        xzb, tb, P1f, Mland, Mland, Mland, Mland, Mland, Mland,
        1, sq, 0, sq, 0, sq, 0);                       // u = xz @ (7I-xz)
    pinv_combine<<<dim3(BHn * 256), 256, 0, stream>>>(P1f, (unsigned short*)nullptr, tb, 15.f);
    gemm_wmma<false, false><<<dim3(4, 4, BHn), 128, 0, stream>>>(
        xzb, tb, P1f, Mland, Mland, Mland, Mland, Mland, Mland,
        1, sq, 0, sq, 0, sq, 0);                       // w = xz @ (15I-u)
    pinv_combine<<<dim3(BHn * 256), 256, 0, stream>>>(P1f, (unsigned short*)nullptr, tb, 13.f);
    gemm_wmma<false, false><<<dim3(4, 4, BHn), 128, 0, stream>>>(
        zb, tb, P0f, Mland, Mland, Mland, Mland, Mland, Mland,
        1, sq, 0, sq, 0, sq, 0);                       // znew = z @ (13I-w)
    scale_to_bf16<<<dim3(BHn * 256), 256, 0, stream>>>(P0f, zb, 0.25f);
  }
  transpose256<<<dim3(BHn * 256), 256, 0, stream>>>(zb, zTb);

  // 8) sim1 = q @ kl^T  [8192,256] per (b,h)
  gemm_wmma<true, true><<<dim3(4, Ntok / 64, BHn), 128, 0, stream>>>(
      qkvb, klb, big1, Ntok, Mland, HDd, QKV3, HDd, Mland,
      Hn, (long long)Ntok * QKV3, HDd,
      (long long)Hn * Mland * HDd, (long long)Mland * HDd,
      (long long)Hn * Ntok * Mland, (long long)Ntok * Mland);
  // attn1 = softmax(scale * sim1)
  softmax_rows<<<dim3((unsigned)(BHn * Ntok)), 256, 0, stream>>>(
      big1, big2, (float*)nullptr, Mland, SCALE);
  // T1 = attn1 @ z   (use zT so K is contiguous)
  gemm_wmma<true, true><<<dim3(4, Ntok / 64, BHn), 128, 0, stream>>>(
      big2, zTb, big3, Ntok, Mland, Mland, Mland, Mland, Mland,
      1, (long long)Ntok * Mland, 0, sq, 0, (long long)Ntok * Mland, 0);

  // 9) sim3 = ql @ k^T  [256,8192] per (b,h)  (reuse big1)
  gemm_wmma<true, true><<<dim3(Ntok / 64, 4, BHn), 128, 0, stream>>>(
      qlb, qkvb + INNER, big1, Mland, Ntok, HDd, HDd, QKV3, Ntok,
      Hn, (long long)Hn * Mland * HDd, (long long)Mland * HDd,
      (long long)Ntok * QKV3, HDd,
      (long long)Hn * Mland * Ntok, (long long)Mland * Ntok);
  // attn3 = softmax(scale * sim3) (reuse big2)
  softmax_rows<<<dim3(BHn * Mland), 256, 0, stream>>>(
      big1, big2, (float*)nullptr, Ntok, SCALE);
  // W = attn3 @ v   [256,64] per (b,h)
  gemm_wmma<false, false><<<dim3(1, 4, BHn), 128, 0, stream>>>(
      big2, qkvb + 2 * INNER, Wf, Mland, HDd, Ntok, Ntok, QKV3, HDd,
      Hn, (long long)Hn * Mland * Ntok, (long long)Mland * Ntok,
      (long long)Ntok * QKV3, HDd,
      (long long)Hn * Mland * HDd, (long long)Mland * HDd);
  wtranspose<<<dim3(BHn * HDd), 256, 0, stream>>>(Wf, WTb);

  // 10) out_heads = T1 @ W  -> [b, n, 512] with per-head column offset
  gemm_wmma<true, true><<<dim3(1, Ntok / 64, BHn), 128, 0, stream>>>(
      big3, WTb, ohb, Ntok, HDd, Mland, Mland, Mland, INNER,
      Hn, (long long)Hn * Ntok * Mland, (long long)Ntok * Mland,
      (long long)Hn * HDd * Mland, (long long)HDd * Mland,
      (long long)Ntok * INNER, HDd);

  // 11) logits = out_heads @ wout^T  (N tiled to 1024, stores clipped to 1000)
  gemm_wmma<true, false><<<dim3(NCLSP / 64, (unsigned)(BN / 64), 1), 128, 0, stream>>>(
      ohb, woutb, out, (int)BN, NCLS, INNER, INNER, INNER, NCLS,
      1, 0, 0, 0, 0, 0, 0);
}
